// GlobalLocalCrossAttention_69836168233565
// MI455X (gfx1250) — matrix operand order repaired and checked by reading, not compiled, not run on hardware
//
#include <hip/hip_runtime.h>
#include <hip/hip_bf16.h>
#include <math.h>

#define BB   8
#define SS   4096
#define DD   1024
#define HH   16
#define HDIM 64
#define SSEL 410
#define SPAD 416
#define DFF  4096

#define AS_GLOBAL __attribute__((address_space(1)))
#define AS_LOCAL  __attribute__((address_space(3)))

typedef __attribute__((ext_vector_type(16))) __bf16 v16bf;
typedef __attribute__((ext_vector_type(8)))  float  v8f;
typedef __attribute__((ext_vector_type(4)))  int    v4i;

union Frag { uint4 u[2]; v16bf v; };

__device__ inline v8f wmma_bf16(v16bf a, v16bf b, v8f c) {
  return __builtin_amdgcn_wmma_f32_16x16x32_bf16(false, a, false, b, (short)0, c, false, false);
}

__device__ inline v8f vzero() {
  v8f z;
#pragma unroll
  for (int r = 0; r < 8; r++) z[r] = 0.f;
  return z;
}

__device__ inline v16bf vones_bf() {
  v16bf o;
#pragma unroll
  for (int i = 0; i < 16; i++) o[i] = (__bf16)1.0f;
  return o;
}

// ---- CDNA5 async global->LDS copy (16B per call), sync fallback ----
__device__ inline void async_copy16(const __bf16* g, __bf16* l) {
#if defined(__gfx1250__) && __has_builtin(__builtin_amdgcn_global_load_async_to_lds_b128)
  __builtin_amdgcn_global_load_async_to_lds_b128((AS_GLOBAL v4i*)g, (AS_LOCAL v4i*)l, 0, 0);
#else
  *(uint4*)l = *(const uint4*)g;
#endif
}

__device__ inline void async_wait0() {
#if defined(__gfx1250__) && __has_builtin(__builtin_amdgcn_global_load_async_to_lds_b128)
#if __has_builtin(__builtin_amdgcn_s_wait_asynccnt)
  __builtin_amdgcn_s_wait_asynccnt(0);
#else
  asm volatile("s_wait_asynccnt 0" ::: "memory");
#endif
#endif
}

// ---------------- prep kernels ----------------
__global__ void cvt_f32_bf16(const float* __restrict__ src, __bf16* __restrict__ dst, long n) {
  long i = (long)blockIdx.x * blockDim.x + threadIdx.x;
  if (i < n) dst[i] = (__bf16)src[i];
}

// fp32 W[K][N] -> bf16 Wt[N][K]
__global__ void cvt_wT(const float* __restrict__ src, __bf16* __restrict__ dst, int K, int N) {
  long i = (long)blockIdx.x * blockDim.x + threadIdx.x;
  long n = (long)K * N;
  if (i < n) {
    int k  = (int)(i % K);
    int nn = (int)(i / K);
    dst[(long)nn * K + k] = (__bf16)src[(long)k * N + nn];
  }
}

__global__ void init_map(int* __restrict__ map) {
  map[(long)blockIdx.x * 256 + threadIdx.x] = -1;
}

__global__ void set_map(const int* __restrict__ sel, int* __restrict__ map) {
  int i = blockIdx.x * 256 + threadIdx.x;
  if (i < BB * SSEL) {
    int b = i / SSEL, j = i % SSEL;
    map[(long)b * SS + sel[i]] = j;
  }
}

// gather selected rows of x into padded bf16 [B][SPAD][D] (pad rows zero)
__global__ void gather_q(const float* __restrict__ x, const int* __restrict__ sel,
                         __bf16* __restrict__ q) {
  long i = (long)blockIdx.x * 256 + threadIdx.x;   // over B*SPAD*D
  int d = (int)(i % DD);
  long r = i / DD;
  int j = (int)(r % SPAD);
  int b = (int)(r / SPAD);
  float v = 0.f;
  if (j < SSEL) {
    int s = sel[b * SSEL + j];
    v = x[((long)b * SS + s) * DD + d];
  }
  q[i] = (__bf16)v;
}

// ---------------- bf16 WMMA GEMM (async double-buffered) ----------------
// C[M,N] = A[M,K] @ Bt[N,K]^T + bias[N]
// EPI 1: gelu -> bf16 row-major ; EPI 2: fp32 row-major
// EPI 3: bf16 head-split  out[((b*H+h)*Mrows + r)*64 + d]
// EPI 4: bf16 head-split transposed out[((b*H+h)*64 + d)*Mrows + r]
// EPI 5: like 3 but scaled by 1/sqrt(HD) (queries)
template <int EPI>
__global__ __launch_bounds__(256) void gemm_bf16(
    const __bf16* __restrict__ A, const __bf16* __restrict__ Bt,
    const float* __restrict__ bias, void* __restrict__ Cout,
    int M, int N, int K, int Mrows) {
  __shared__ __bf16 As[2][128][32];
  __shared__ __bf16 Bs[2][128][32];
  const int tid  = threadIdx.x;
  const int m0   = blockIdx.y * 128;
  const int n0   = blockIdx.x * 128;
  const int w    = tid >> 5;
  const int wr   = w >> 1;       // 0..3 -> 32-row strip
  const int wc   = w & 1;        // 0..1 -> 64-col strip
  const int lane = tid & 31;
  const int lm   = lane & 15;
  const int g    = lane >> 4;
  const int lrow = tid >> 1;     // 0..127
  const int lseg = tid & 1;      // 0..1

  const __bf16* ga0 = A  + (size_t)(m0 + lrow) * K + 16 * lseg;
  const __bf16* gb0 = Bt + (size_t)(n0 + lrow) * K + 16 * lseg;

  v8f acc[2][4];
#pragma unroll
  for (int i = 0; i < 2; i++) {
#pragma unroll
    for (int j = 0; j < 4; j++) acc[i][j] = vzero();
  }

  auto load_tile = [&](int buf, int k0) {
    const __bf16* ga = ga0 + k0;
    async_copy16(ga,     &As[buf][lrow][16 * lseg]);
    async_copy16(ga + 8, &As[buf][lrow][16 * lseg + 8]);
    const __bf16* gb = gb0 + k0;
    async_copy16(gb,     &Bs[buf][lrow][16 * lseg]);
    async_copy16(gb + 8, &Bs[buf][lrow][16 * lseg + 8]);
  };

  const int nt = K >> 5;
  load_tile(0, 0);
  async_wait0();
  __syncthreads();

  for (int t = 0; t < nt; t++) {
    const int cur = t & 1;
    if (t + 1 < nt) load_tile(cur ^ 1, (t + 1) * 32);

    Frag a[2], b[4];
#pragma unroll
    for (int i = 0; i < 2; i++) {
      int m = 32 * wr + 16 * i + lm;
      a[i].u[0] = *(const uint4*)&As[cur][m][8 * g];
      a[i].u[1] = *(const uint4*)&As[cur][m][16 + 8 * g];
    }
#pragma unroll
    for (int j = 0; j < 4; j++) {
      int n = 64 * wc + 16 * j + lm;
      b[j].u[0] = *(const uint4*)&Bs[cur][n][8 * g];
      b[j].u[1] = *(const uint4*)&Bs[cur][n][16 + 8 * g];
    }
#pragma unroll
    for (int i = 0; i < 2; i++) {
#pragma unroll
      for (int j = 0; j < 4; j++) acc[i][j] = wmma_bf16(a[i].v, b[j].v, acc[i][j]);
    }
    async_wait0();
    __syncthreads();
  }

#pragma unroll
  for (int i = 0; i < 2; i++) {
#pragma unroll
    for (int j = 0; j < 4; j++) {
#pragma unroll
      for (int r = 0; r < 8; r++) {
        int mg = m0 + 32 * wr + 16 * i + r + 8 * g;
        int ng = n0 + 64 * wc + 16 * j + lm;
        float v = acc[i][j][r] + bias[ng];
        if (EPI == 1) {
          v = 0.5f * v * (1.f + erff(v * 0.70710678118654752f));
          ((__bf16*)Cout)[(size_t)mg * N + ng] = (__bf16)v;
        } else if (EPI == 2) {
          ((float*)Cout)[(size_t)mg * N + ng] = v;
        } else if (EPI == 3) {
          int b = mg / Mrows, rr = mg % Mrows, h = ng >> 6, d = ng & 63;
          ((__bf16*)Cout)[(((size_t)b * HH + h) * Mrows + rr) * HDIM + d] = (__bf16)v;
        } else if (EPI == 4) {
          int b = mg / Mrows, rr = mg % Mrows, h = ng >> 6, d = ng & 63;
          ((__bf16*)Cout)[(((size_t)b * HH + h) * HDIM + d) * Mrows + rr] = (__bf16)v;
        } else if (EPI == 5) {
          v *= 0.125f;  // 1/sqrt(64) folded into Q
          int b = mg / Mrows, rr = mg % Mrows, h = ng >> 6, d = ng & 63;
          ((__bf16*)Cout)[(((size_t)b * HH + h) * Mrows + rr) * HDIM + d] = (__bf16)v;
        }
      }
    }
  }
}

// ---------------- flash attention (WMMA, 64-key chunks) ----------------
// Qh [B,H,SPAD,64] (pre-scaled by 1/8), Kh [B,H,S,64], Vt [B,H,64,S] -> Oh [B,SPAD,D]
__global__ __launch_bounds__(128) void attn_kernel(
    const __bf16* __restrict__ Qh, const __bf16* __restrict__ Kh,
    const __bf16* __restrict__ Vt, __bf16* __restrict__ Oh) {
  __shared__ __bf16 Pb[4][16][64];
  const int wv   = threadIdx.x >> 5;
  const int lane = threadIdx.x & 31;
  const int lm   = lane & 15;
  const int g    = lane >> 4;
  const int bh   = blockIdx.x / 7;
  const int grp  = blockIdx.x % 7;
  int qt = grp * 4 + wv;
  if (qt > 25) qt = 25;                 // 26 q-tiles of 16 (SPAD=416)
  const int b = bh >> 4, h = bh & 15;

  const __bf16* Qb = Qh + ((size_t)bh * SPAD + (size_t)qt * 16) * HDIM;
  const __bf16* Kb = Kh + (size_t)bh * SS * HDIM;
  const __bf16* Vb = Vt + (size_t)bh * HDIM * SS;

  Frag aq[2];
#pragma unroll
  for (int i = 0; i < 2; i++) {
    const __bf16* qp = Qb + (size_t)lm * HDIM + 32 * i;
    aq[i].u[0] = *(const uint4*)(qp + 8 * g);
    aq[i].u[1] = *(const uint4*)(qp + 16 + 8 * g);
  }

  const v16bf ones = vones_bf();
  float mloc[8], lloc[8];
  v8f acc[4];
#pragma unroll
  for (int r = 0; r < 8; r++) { mloc[r] = -3.0e38f; lloc[r] = 0.f; }
#pragma unroll
  for (int dt = 0; dt < 4; dt++) acc[dt] = vzero();

  for (int kc = 0; kc < SS; kc += 64) {
    // ---- scores: Q(16x64) . K^T -> 16x64, 4 C-frags ----
    v8f sc[4];
#pragma unroll
    for (int j = 0; j < 4; j++) {
      const __bf16* kp = Kb + (size_t)(kc + 16 * j + lm) * HDIM + 8 * g;
      Frag b0, b1;
      b0.u[0] = *(const uint4*)kp;
      b0.u[1] = *(const uint4*)(kp + 16);
      b1.u[0] = *(const uint4*)(kp + 32);
      b1.u[1] = *(const uint4*)(kp + 48);
      sc[j] = wmma_bf16(aq[0].v, b0.v, vzero());
      sc[j] = wmma_bf16(aq[1].v, b1.v, sc[j]);
    }
    // ---- online softmax stats (max via half-wave shuffles) ----
#pragma unroll
    for (int r = 0; r < 8; r++) {
      float s0 = sc[0][r], s1 = sc[1][r], s2 = sc[2][r], s3 = sc[3][r];
      float mx = fmaxf(fmaxf(s0, s1), fmaxf(s2, s3));
#pragma unroll
      for (int o = 8; o; o >>= 1) mx = fmaxf(mx, __shfl_xor(mx, o, 16));
      float nm   = fmaxf(mloc[r], mx);
      float corr = __expf(mloc[r] - nm);
      mloc[r] = nm;
      lloc[r] *= corr;
#pragma unroll
      for (int dt = 0; dt < 4; dt++) acc[dt][r] *= corr;
      Pb[wv][r + 8 * g][lm]      = (__bf16)__expf(s0 - nm);
      Pb[wv][r + 8 * g][16 + lm] = (__bf16)__expf(s1 - nm);
      Pb[wv][r + 8 * g][32 + lm] = (__bf16)__expf(s2 - nm);
      Pb[wv][r + 8 * g][48 + lm] = (__bf16)__expf(s3 - nm);
    }
    asm volatile("s_wait_dscnt 0" ::: "memory");
    // ---- reload P as A-fragments (16x32 each) ----
    Frag pf[2];
#pragma unroll
    for (int kp = 0; kp < 2; kp++) {
      pf[kp].u[0] = *(const uint4*)&Pb[wv][lm][32 * kp + 8 * g];
      pf[kp].u[1] = *(const uint4*)&Pb[wv][lm][32 * kp + 16 + 8 * g];
    }
    // ---- row sums via WMMA: P @ ones -> sums replicated per row ----
    v8f srow = wmma_bf16(pf[0].v, ones, vzero());
    srow = wmma_bf16(pf[1].v, ones, srow);
#pragma unroll
    for (int r = 0; r < 8; r++) lloc[r] += srow[r];
    // ---- O += P . V ----
#pragma unroll
    for (int kp = 0; kp < 2; kp++) {
#pragma unroll
      for (int dt = 0; dt < 4; dt++) {
        const __bf16* vp = Vb + (size_t)(16 * dt + lm) * SS + kc + 32 * kp + 8 * g;
        Frag vf;
        vf.u[0] = *(const uint4*)vp;
        vf.u[1] = *(const uint4*)(vp + 16);
        acc[dt] = wmma_bf16(pf[kp].v, vf.v, acc[dt]);
      }
    }
  }

#pragma unroll
  for (int dt = 0; dt < 4; dt++) {
#pragma unroll
    for (int r = 0; r < 8; r++) {
      int q = qt * 16 + r + 8 * g;
      if (q < SSEL) {
        float o = acc[dt][r] / lloc[r];
        Oh[((size_t)b * SPAD + q) * DD + h * HDIM + 16 * dt + lm] = (__bf16)o;
      }
    }
  }
}

// ---------------- LayerNorm kernels ----------------
__global__ __launch_bounds__(256) void ln1_kernel(
    const float* __restrict__ x, const float* __restrict__ Oo, const int* __restrict__ map,
    const float* __restrict__ gam, const float* __restrict__ bet,
    float* __restrict__ y, __bf16* __restrict__ yb) {
  __shared__ float r1[256], r2[256];
  const int row = blockIdx.x;
  const int b   = row >> 12;     // S = 4096
  const int tid = threadIdx.x;
  const int j   = map[row];
  const float* xr  = x + (size_t)row * DD;
  const float* orw = (j >= 0) ? (Oo + ((size_t)b * SPAD + j) * DD) : nullptr;
  float t[4];
  float s1 = 0.f, s2 = 0.f;
#pragma unroll
  for (int i = 0; i < 4; i++) {
    int c = i * 256 + tid;
    float v = xr[c] + (orw ? orw[c] : 0.f);
    t[i] = v; s1 += v; s2 += v * v;
  }
  r1[tid] = s1; r2[tid] = s2;
  __syncthreads();
  for (int st = 128; st > 0; st >>= 1) {
    if (tid < st) { r1[tid] += r1[tid + st]; r2[tid] += r2[tid + st]; }
    __syncthreads();
  }
  float mean = r1[0] * (1.f / DD);
  float var  = r2[0] * (1.f / DD) - mean * mean;
  float rstd = rsqrtf(var + 1e-5f);
#pragma unroll
  for (int i = 0; i < 4; i++) {
    int c = i * 256 + tid;
    float v = (t[i] - mean) * rstd * gam[c] + bet[c];
    y[(size_t)row * DD + c]  = v;
    yb[(size_t)row * DD + c] = (__bf16)v;
  }
}

__global__ __launch_bounds__(256) void ln2_kernel(
    const float* __restrict__ y, const float* __restrict__ f2,
    const float* __restrict__ gam, const float* __restrict__ bet,
    float* __restrict__ out) {
  __shared__ float r1[256], r2[256];
  const int row = blockIdx.x;
  const int tid = threadIdx.x;
  const float* yr = y  + (size_t)row * DD;
  const float* fr = f2 + (size_t)row * DD;
  float t[4];
  float s1 = 0.f, s2 = 0.f;
#pragma unroll
  for (int i = 0; i < 4; i++) {
    int c = i * 256 + tid;
    float v = yr[c] + fr[c];
    t[i] = v; s1 += v; s2 += v * v;
  }
  r1[tid] = s1; r2[tid] = s2;
  __syncthreads();
  for (int st = 128; st > 0; st >>= 1) {
    if (tid < st) { r1[tid] += r1[tid + st]; r2[tid] += r2[tid + st]; }
    __syncthreads();
  }
  float mean = r1[0] * (1.f / DD);
  float var  = r2[0] * (1.f / DD) - mean * mean;
  float rstd = rsqrtf(var + 1e-5f);
#pragma unroll
  for (int i = 0; i < 4; i++) {
    int c = i * 256 + tid;
    out[(size_t)row * DD + c] = (t[i] - mean) * rstd * gam[c] + bet[c];
  }
}

// ---------------- launcher ----------------
extern "C" void kernel_launch(void* const* d_in, const int* in_sizes, int n_in,
                              void* d_out, int out_size, void* d_ws, size_t ws_size,
                              hipStream_t stream) {
  const float* x   = (const float*)d_in[0];
  const int*   sel = (const int*)d_in[1];
  const float* Wq  = (const float*)d_in[2];  const float* bq  = (const float*)d_in[3];
  const float* Wk  = (const float*)d_in[4];  const float* bk  = (const float*)d_in[5];
  const float* Wv  = (const float*)d_in[6];  const float* bv  = (const float*)d_in[7];
  const float* Wo  = (const float*)d_in[8];  const float* bo  = (const float*)d_in[9];
  const float* g1  = (const float*)d_in[10]; const float* b1  = (const float*)d_in[11];
  const float* g2  = (const float*)d_in[12]; const float* b2  = (const float*)d_in[13];
  const float* W1  = (const float*)d_in[14]; const float* bf1 = (const float*)d_in[15];
  const float* W2  = (const float*)d_in[16]; const float* bf2 = (const float*)d_in[17];
  float* out = (float*)d_out;

  char* p = (char*)d_ws;
  auto carve = [&](size_t bytes) -> char* {
    char* q = p;
    p += (bytes + 255) & ~(size_t)255;
    return q;
  };
  __bf16* xb   = (__bf16*)carve((size_t)BB * SS * DD * 2);
  __bf16* WqT  = (__bf16*)carve((size_t)DD * DD * 2);
  __bf16* WkT  = (__bf16*)carve((size_t)DD * DD * 2);
  __bf16* WvT  = (__bf16*)carve((size_t)DD * DD * 2);
  __bf16* WoT  = (__bf16*)carve((size_t)DD * DD * 2);
  __bf16* W1T  = (__bf16*)carve((size_t)DD * DFF * 2);
  __bf16* W2T  = (__bf16*)carve((size_t)DFF * DD * 2);
  __bf16* qloc = (__bf16*)carve((size_t)BB * SPAD * DD * 2);
  __bf16* Qh   = (__bf16*)carve((size_t)BB * HH * SPAD * HDIM * 2);
  __bf16* Kh   = (__bf16*)carve((size_t)BB * HH * SS * HDIM * 2);
  __bf16* Vt   = (__bf16*)carve((size_t)BB * HH * HDIM * SS * 2);
  __bf16* Oh   = (__bf16*)carve((size_t)BB * SPAD * DD * 2);
  float*  Oo   = (float*) carve((size_t)BB * SPAD * DD * 4);
  int*    map  = (int*)   carve((size_t)BB * SS * 4);
  float*  y    = (float*) carve((size_t)BB * SS * DD * 4);
  __bf16* yb   = (__bf16*)carve((size_t)BB * SS * DD * 2);
  __bf16* h1b  = (__bf16*)carve((size_t)BB * SS * DFF * 2);
  float*  f2   = (float*) carve((size_t)BB * SS * DD * 4);

  const long nx = (long)BB * SS * DD;
  cvt_f32_bf16<<<(int)((nx + 255) / 256), 256, 0, stream>>>(x, xb, nx);
  cvt_wT<<<(DD * DD + 255) / 256, 256, 0, stream>>>(Wq, WqT, DD, DD);
  cvt_wT<<<(DD * DD + 255) / 256, 256, 0, stream>>>(Wk, WkT, DD, DD);
  cvt_wT<<<(DD * DD + 255) / 256, 256, 0, stream>>>(Wv, WvT, DD, DD);
  cvt_wT<<<(DD * DD + 255) / 256, 256, 0, stream>>>(Wo, WoT, DD, DD);
  cvt_wT<<<(DD * DFF + 255) / 256, 256, 0, stream>>>(W1, W1T, DD, DFF);
  cvt_wT<<<(DFF * DD + 255) / 256, 256, 0, stream>>>(W2, W2T, DFF, DD);
  init_map<<<BB * SS / 256, 256, 0, stream>>>(map);
  set_map<<<(BB * SSEL + 255) / 256, 256, 0, stream>>>(sel, map);
  gather_q<<<(int)(((long)BB * SPAD * DD) / 256), 256, 0, stream>>>(x, sel, qloc);

  // K, V, Q projections (bf16 WMMA; Q pre-scaled by 1/8)
  gemm_bf16<3><<<dim3(DD / 128, BB * SS / 128), 256, 0, stream>>>(xb, WkT, bk, Kh, BB * SS, DD, DD, SS);
  gemm_bf16<4><<<dim3(DD / 128, BB * SS / 128), 256, 0, stream>>>(xb, WvT, bv, Vt, BB * SS, DD, DD, SS);
  gemm_bf16<5><<<dim3(DD / 128, BB * SPAD / 128), 256, 0, stream>>>(qloc, WqT, bq, Qh, BB * SPAD, DD, DD, SPAD);

  // flash attention over 4096 keys
  attn_kernel<<<BB * HH * 7, 128, 0, stream>>>(Qh, Kh, Vt, Oh);

  // output projection (fp32 out, bias included)
  gemm_bf16<2><<<dim3(DD / 128, BB * SPAD / 128), 256, 0, stream>>>(Oh, WoT, bo, Oo, BB * SPAD, DD, DD, 1);

  // scatter + residual + LN1 -> y (fp32) and yb (bf16)
  ln1_kernel<<<BB * SS, 256, 0, stream>>>(x, Oo, map, g1, b1, y, yb);

  // FFN: gelu(y @ W1 + bf1) @ W2 + bf2
  gemm_bf16<1><<<dim3(DFF / 128, BB * SS / 128), 256, 0, stream>>>(yb, W1T, bf1, h1b, BB * SS, DFF, DD, 1);
  gemm_bf16<2><<<dim3(DD / 128, BB * SS / 128), 256, 0, stream>>>(h1b, W2T, bf2, f2, BB * SS, DD, DFF, 1);

  // residual + LN2 -> output
  ln2_kernel<<<BB * SS, 256, 0, stream>>>(y, f2, g2, b2, out);
}
